// GNNEncoder_14156212208481
// MI455X (gfx1250) — matrix operand order repaired and checked by reading, not compiled, not run on hardware
//
#include <hip/hip_runtime.h>
#include <hip/hip_bf16.h>

// ---------------------------------------------------------------------------
// 3-layer GAT (PyG GATConv semantics) for MI455X / gfx1250.
//   N=100000 nodes (multiple of 16), IN=HID=OUT=128, layers 0/1: H=4,C=32,
//   layer 2: H=1,C=128. Self-loops appended (Etot = E + N).
// Dense projection g = h @ W runs on v_wmma_f32_16x16x32_bf16 (16x16 tiles,
// K=128 in 4 steps of 32). Edge softmax/aggregation via float atomics.
// ---------------------------------------------------------------------------

typedef __attribute__((ext_vector_type(16))) __bf16 v16bf;
typedef __attribute__((ext_vector_type(8)))  float  v8f;

#define FDIM 128   // feature dim of every layer in/out

// ---- order-preserving float<->uint encoding for atomic max --------------
__device__ __forceinline__ unsigned int encf(float f) {
    unsigned int u = __float_as_uint(f);
    return (u & 0x80000000u) ? ~u : (u | 0x80000000u);
}
__device__ __forceinline__ float decf(unsigned int u) {
    return (u & 0x80000000u) ? __uint_as_float(u & 0x7FFFFFFFu)
                             : __uint_as_float(~u);
}

// ---------------------------------------------------------------------------
// zero fill
// ---------------------------------------------------------------------------
__global__ void zero_f32(float* __restrict__ p, int n) {
    int i = blockIdx.x * blockDim.x + threadIdx.x;
    if (i < n) p[i] = 0.0f;
}

// ---------------------------------------------------------------------------
// WMMA GEMM: G[M,128] = A[M,128] @ W[128,128]   (M multiple of 16)
// Block = 256 threads (8 waves). Wave w handles row-tile blockIdx.x*8+w.
// W is staged transposed in LDS as bf16 so each B fragment is a contiguous
// 32-byte LDS vector load per lane.
// ---------------------------------------------------------------------------
__global__ __launch_bounds__(256)
void gat_gemm_wmma(const float* __restrict__ A,
                   const float* __restrict__ W,
                   float* __restrict__ G,
                   int Mtiles) {
    __shared__ __bf16 WT[FDIM * FDIM];          // W transposed: WT[n][k], 32 KB

    const int t = threadIdx.x;
    for (int i = t; i < FDIM * FDIM; i += 256) {
        int k = i >> 7;          // row of W
        int n = i & 127;         // col of W
        WT[n * FDIM + k] = (__bf16)W[i];
    }
    __syncthreads();

    const int wave = t >> 5;
    const int lane = t & 31;
    const int rowTile = blockIdx.x * 8 + wave;
    if (rowTile >= Mtiles) return;               // wave-uniform: EXEC stays all-1s

    const int m16 = lane & 15;                   // row within tile (A) / col (B,C)
    const int hi  = lane >> 4;                   // lane-half selector
    const int rowBase = rowTile * 16;

    // ---- preload the 4 A fragments (one per K-step of 32) ----------------
    // A 16x32 layout (matches B): lane<16 holds K {kb..kb+15},
    // lane>=16 holds K {kb+16..kb+31}.
    const float* Arow = A + (size_t)(rowBase + m16) * FDIM;
    v16bf afrag[4];
#pragma unroll
    for (int ks = 0; ks < 4; ++ks) {
        const int kb = ks * 32 + hi * 16;
        float4 a0 = *(const float4*)(Arow + kb);
        float4 a1 = *(const float4*)(Arow + kb + 4);
        float4 a2 = *(const float4*)(Arow + kb + 8);
        float4 a3 = *(const float4*)(Arow + kb + 12);
        v16bf f;
        f[0]  = (__bf16)a0.x; f[1]  = (__bf16)a0.y; f[2]  = (__bf16)a0.z; f[3]  = (__bf16)a0.w;
        f[4]  = (__bf16)a1.x; f[5]  = (__bf16)a1.y; f[6]  = (__bf16)a1.z; f[7]  = (__bf16)a1.w;
        f[8]  = (__bf16)a2.x; f[9]  = (__bf16)a2.y; f[10] = (__bf16)a2.z; f[11] = (__bf16)a2.w;
        f[12] = (__bf16)a3.x; f[13] = (__bf16)a3.y; f[14] = (__bf16)a3.z; f[15] = (__bf16)a3.w;
        afrag[ks] = f;
    }

    // ---- 8 column tiles, 4 WMMA (K=32) each ------------------------------
#pragma unroll
    for (int ct = 0; ct < 8; ++ct) {
        const int ncol = ct * 16 + m16;          // this lane's output column
        v8f acc = {};
#pragma unroll
        for (int ks = 0; ks < 4; ++ks) {
            // B 32x16 layout: lane<16 holds K 0..15, lane>=16 holds K 16..31
            const int kb = ks * 32 + hi * 16;
            v16bf bfrag = *(const v16bf*)(&WT[ncol * FDIM + kb]);
            acc = __builtin_amdgcn_wmma_f32_16x16x32_bf16(
                      false, afrag[ks], false, bfrag,
                      (short)0, acc, false, false);
        }
        // C/D layout: lane holds column ncol, rows rowBase + hi*8 + r
        float* gp = G + (size_t)(rowBase + hi * 8) * FDIM + ncol;
#pragma unroll
        for (int r = 0; r < 8; ++r) gp[(size_t)r * FDIM] = acc[r];
    }
}

// ---------------------------------------------------------------------------
// al[n,h] = sum_c g[n,h,c]*a_src[h,c];  ar likewise with a_dst
// ---------------------------------------------------------------------------
__global__ void gat_attn(const float* __restrict__ G,
                         const float* __restrict__ a_src,
                         const float* __restrict__ a_dst,
                         float* __restrict__ al, float* __restrict__ ar,
                         int N, int HEADS, int C) {
    int idx = blockIdx.x * blockDim.x + threadIdx.x;
    if (idx >= N * HEADS) return;
    int n = idx / HEADS, h = idx - n * HEADS;
    const float* g  = G + (size_t)n * FDIM + h * C;
    const float* as = a_src + h * C;
    const float* ad = a_dst + h * C;
    float sl = 0.f, sr = 0.f;
    for (int c = 0; c < C; c += 4) {
        float4 gv = *(const float4*)(g + c);
        float4 sv = *(const float4*)(as + c);
        float4 dv = *(const float4*)(ad + c);
        sl += gv.x * sv.x + gv.y * sv.y + gv.z * sv.z + gv.w * sv.w;
        sr += gv.x * dv.x + gv.y * dv.y + gv.z * dv.z + gv.w * dv.w;
    }
    al[idx] = sl;
    ar[idx] = sr;
}

// ---------------------------------------------------------------------------
// pass 1: sc = leaky_relu(al[src]+ar[dst], 0.2); segment-max into m[dst,h]
// ---------------------------------------------------------------------------
__global__ void gat_edge_max(const int* __restrict__ ei, int E, int Etot, int HEADS,
                             const float* __restrict__ al, const float* __restrict__ ar,
                             float* __restrict__ sc, unsigned int* __restrict__ m) {
    int idx = blockIdx.x * blockDim.x + threadIdx.x;
    if (idx >= Etot * HEADS) return;
    int e = idx / HEADS, h = idx - e * HEADS;
    int s, d;
    if (e < E) { s = ei[e]; d = ei[E + e]; } else { s = d = e - E; }  // self loops
    float v = al[s * HEADS + h] + ar[d * HEADS + h];
    v = v > 0.f ? v : 0.2f * v;
    sc[idx] = v;
    atomicMax(&m[d * HEADS + h], encf(v));
}

// ---------------------------------------------------------------------------
// pass 2: ex = exp(sc - m[dst]); segment-sum into s[dst,h]; sc <- ex in place
// ---------------------------------------------------------------------------
__global__ void gat_edge_exp(const int* __restrict__ ei, int E, int Etot, int HEADS,
                             const unsigned int* __restrict__ m,
                             float* __restrict__ sc, float* __restrict__ ssum) {
    int idx = blockIdx.x * blockDim.x + threadIdx.x;
    if (idx >= Etot * HEADS) return;
    int e = idx / HEADS, h = idx - e * HEADS;
    int d = (e < E) ? ei[E + e] : (e - E);
    float ex = __expf(sc[idx] - decf(m[d * HEADS + h]));
    sc[idx] = ex;
    atomicAdd(&ssum[d * HEADS + h], ex);
}

// ---------------------------------------------------------------------------
// pass 3: acc[dst, :] += g[src, :] * (ex/(s[dst]+eps)).  One wave per edge;
// lane handles 4 consecutive channels (float4 gather + 4 float atomics).
// ---------------------------------------------------------------------------
__global__ __launch_bounds__(256)
void gat_edge_aggr(const int* __restrict__ ei, int E, int Etot, int HEADS, int C,
                   const float* __restrict__ ex, const float* __restrict__ ssum,
                   const float* __restrict__ G, float* __restrict__ acc) {
    int wave = blockIdx.x * 8 + (threadIdx.x >> 5);
    int lane = threadIdx.x & 31;
    if (wave >= Etot) return;
    int e = wave;
    int s, d;
    if (e < E) { s = ei[e]; d = ei[E + e]; } else { s = d = e - E; }
    int c0 = lane * 4;                 // 4 channels per lane, same head (C%4==0)
    int h  = c0 / C;
    float alpha = ex[(size_t)e * HEADS + h] / (ssum[d * HEADS + h] + 1e-16f);
    float4 gv = *(const float4*)(G + (size_t)s * FDIM + c0);
    float* ap = acc + (size_t)d * FDIM + c0;
    atomicAdd(ap + 0, gv.x * alpha);
    atomicAdd(ap + 1, gv.y * alpha);
    atomicAdd(ap + 2, gv.z * alpha);
    atomicAdd(ap + 3, gv.w * alpha);
}

// ---------------------------------------------------------------------------
// out = (relu ? max(0, acc+b) : acc+b), in place allowed
// ---------------------------------------------------------------------------
__global__ void gat_finalize(const float* __restrict__ acc, const float* __restrict__ b,
                             float* __restrict__ dst, int total, int doRelu) {
    int idx = blockIdx.x * blockDim.x + threadIdx.x;
    if (idx >= total) return;
    float v = acc[idx] + b[idx & (FDIM - 1)];
    if (doRelu) v = fmaxf(v, 0.0f);
    dst[idx] = v;
}

// ---------------------------------------------------------------------------
// host side
// ---------------------------------------------------------------------------
static inline int cdiv(int a, int b) { return (a + b - 1) / b; }

struct Bufs {
    float* g; float* al; float* ar; unsigned int* m; float* s; float* sc;
    float* h1; float* h2;
};

static void run_layer(const float* hin, const float* W, const float* as,
                      const float* ad, const float* b, int HEADS, int C,
                      float* accOut, int doRelu,
                      int N, int E, int Etot, const int* ei,
                      const Bufs& bf, hipStream_t st) {
    // zero accumulators
    zero_f32<<<cdiv(N * FDIM, 256), 256, 0, st>>>(accOut, N * FDIM);
    zero_f32<<<cdiv(N * HEADS, 256), 256, 0, st>>>((float*)bf.m, N * HEADS);
    zero_f32<<<cdiv(N * HEADS, 256), 256, 0, st>>>(bf.s, N * HEADS);

    // g = hin @ W  (WMMA)
    gat_gemm_wmma<<<cdiv(N / 16, 8), 256, 0, st>>>(hin, W, bf.g, N / 16);

    // attention coefficients
    gat_attn<<<cdiv(N * HEADS, 256), 256, 0, st>>>(bf.g, as, ad, bf.al, bf.ar,
                                                   N, HEADS, C);
    // edge softmax + aggregation
    gat_edge_max<<<cdiv(Etot * HEADS, 256), 256, 0, st>>>(ei, E, Etot, HEADS,
                                                          bf.al, bf.ar, bf.sc, bf.m);
    gat_edge_exp<<<cdiv(Etot * HEADS, 256), 256, 0, st>>>(ei, E, Etot, HEADS,
                                                          bf.m, bf.sc, bf.s);
    gat_edge_aggr<<<cdiv(Etot, 8), 256, 0, st>>>(ei, E, Etot, HEADS, C,
                                                 bf.sc, bf.s, bf.g, accOut);
    // bias (+ relu) in place
    gat_finalize<<<cdiv(N * FDIM, 256), 256, 0, st>>>(accOut, b, accOut,
                                                      N * FDIM, doRelu);
}

extern "C" void kernel_launch(void* const* d_in, const int* in_sizes, int n_in,
                              void* d_out, int out_size, void* d_ws, size_t ws_size,
                              hipStream_t stream) {
    const float* x   = (const float*)d_in[0];
    const int*   ei  = (const int*)  d_in[1];     // [2,E]: row0=src, row1=dst
    const float* W0  = (const float*)d_in[2];
    const float* as0 = (const float*)d_in[3];
    const float* ad0 = (const float*)d_in[4];
    const float* b0  = (const float*)d_in[5];
    const float* W1  = (const float*)d_in[6];
    const float* as1 = (const float*)d_in[7];
    const float* ad1 = (const float*)d_in[8];
    const float* b1  = (const float*)d_in[9];
    const float* W2  = (const float*)d_in[10];
    const float* as2 = (const float*)d_in[11];
    const float* ad2 = (const float*)d_in[12];
    const float* b2  = (const float*)d_in[13];

    const int N    = in_sizes[0] / FDIM;   // 100000 (multiple of 16)
    const int E    = in_sizes[1] / 2;      // 1600000
    const int Etot = E + N;                // + self loops
    const int HMAX = 4;

    // workspace carve-up (all fp32-sized, 16B aligned)
    char* w = (char*)d_ws;
    Bufs bf;
    bf.g  = (float*)w;        w += (size_t)N * FDIM * 4;
    bf.al = (float*)w;        w += (size_t)N * HMAX * 4;
    bf.ar = (float*)w;        w += (size_t)N * HMAX * 4;
    bf.m  = (unsigned int*)w; w += (size_t)N * HMAX * 4;
    bf.s  = (float*)w;        w += (size_t)N * HMAX * 4;
    bf.sc = (float*)w;        w += (size_t)Etot * HMAX * 4;
    bf.h1 = (float*)w;        w += (size_t)N * FDIM * 4;
    bf.h2 = (float*)w;        w += (size_t)N * FDIM * 4;

    float* out = (float*)d_out;

    // layer 0: IN=128 -> 4 heads x 32, relu
    run_layer(x,     W0, as0, ad0, b0, 4, 32,  bf.h1, 1, N, E, Etot, ei, bf, stream);
    // layer 1: 128 -> 4 heads x 32, relu
    run_layer(bf.h1, W1, as1, ad1, b1, 4, 32,  bf.h2, 1, N, E, Etot, ei, bf, stream);
    // layer 2: 128 -> 1 head x 128, no relu, straight into d_out
    run_layer(bf.h2, W2, as2, ad2, b2, 1, 128, out,   0, N, E, Etot, ei, bf, stream);
}
